// AttentionWithCache_33251636806034
// MI455X (gfx1250) — matrix operand order repaired and checked by reading, not compiled, not run on hardware
//
#include <hip/hip_runtime.h>
#include <hip/hip_bf16.h>

typedef __attribute__((ext_vector_type(16))) __bf16   v16bf;
typedef __attribute__((ext_vector_type(8)))  float    v8f;
typedef __attribute__((ext_vector_type(4)))  unsigned uint32x4;
typedef __attribute__((ext_vector_type(8)))  unsigned uint32x8;

#define D_MODEL  4096
#define T_SEQ    128
#define HEAD_DIM 128
#define N_CHUNK  4            // flash-decoding S splits

#if __has_builtin(__builtin_amdgcn_sched_barrier)
#define SCHED_FENCE() __builtin_amdgcn_sched_barrier(0)
#else
#define SCHED_FENCE()
#endif

__device__ __forceinline__ void wait_tensorcnt0() {
#if __has_builtin(__builtin_amdgcn_s_wait_tensorcnt)
    __builtin_amdgcn_s_wait_tensorcnt(0);
#else
    asm volatile("s_wait_tensorcnt 0x0" ::: "memory");
#endif
}

// ---------------------------------------------------------------------------
// Issue a 2D TDM load: tile (tileX x tileY) of 4-byte elements from a
// row-major tensor with row stride strideX elements, into LDS at ldsAddr.
// D# packing per CDNA5 ISA 8.3/8.4 (group0 128b + group1 256b, 2D form).
// ---------------------------------------------------------------------------
__device__ __forceinline__ void tdm_load_2d_f32(const float* gaddr, unsigned ldsAddr,
                                                unsigned tileX, unsigned tileY,
                                                unsigned tensorX, unsigned tensorY,
                                                unsigned strideX)
{
    const unsigned long long ga = (unsigned long long)(const void*)gaddr;
    uint32x4 g0;
    g0[0] = 1u;                                          // count=1 (valid user D#)
    g0[1] = ldsAddr;                                     // lds byte address
    g0[2] = (unsigned)(ga & 0xFFFFFFFFu);                // global_addr[31:0]
    g0[3] = (unsigned)((ga >> 32) & 0x01FFFFFFu) | (2u << 30);  // addr[56:32] | type=2
    uint32x8 g1;
    g1[0] = (2u << 16);                                  // wg_mask=0, data_size=4B
    g1[1] = (tensorX & 0xFFFFu) << 16;                   // tensor_dim0[15:0] @bit48
    g1[2] = ((tensorX >> 16) & 0xFFFFu) | ((tensorY & 0xFFFFu) << 16);
    g1[3] = ((tensorY >> 16) & 0xFFFFu) | ((tileX & 0xFFFFu) << 16);  // tile_dim0
    g1[4] = (tileY & 0xFFFFu);                           // tile_dim1 (tile_dim2=0)
    g1[5] = strideX;                                     // tensor_dim0_stride[31:0]
    g1[6] = 0u;                                          // stride hi / dim1_stride lo
    g1[7] = 0u;
    asm volatile("tensor_load_to_lds %0, %1" :: "s"(g0), "s"(g1) : "memory");
}

__device__ __forceinline__ v8f zero8() {
    v8f z = {0.f, 0.f, 0.f, 0.f, 0.f, 0.f, 0.f, 0.f};
    return z;
}

__device__ __forceinline__ float half_max16(float v) {
#pragma unroll
    for (int m = 1; m < 16; m <<= 1) v = fmaxf(v, __shfl_xor(v, m, 32));
    return v;
}
__device__ __forceinline__ float half_sum16(float v) {
#pragma unroll
    for (int m = 1; m < 16; m <<= 1) v += __shfl_xor(v, m, 32);
    return v;
}

// ---------------------------------------------------------------------------
// C[128,4096] = X[128,4096] @ W^T + bias, fp32 in/out, bf16 WMMA compute.
// grid = 64 blocks (N tiles of 64), block = 128 threads = 4 waves.
// X tile arrives via double-buffered TDM (tensor_load_to_lds) as raw fp32;
// a short LDS pass converts it to bf16; W streams via global_load_b128 +
// prefetch (zero reuse -> no staging). A-fragments preload behind a
// sched_barrier so the 8 WMMAs issue as a dense stream.
// ---------------------------------------------------------------------------
__global__ __launch_bounds__(128)
void gemm_xwT_bias(const float* __restrict__ X, const float* __restrict__ W,
                   const float* __restrict__ bias, float* __restrict__ C)
{
    __shared__ __align__(64) float  xraw[2][T_SEQ * 32]; // 2 x 16KB raw TDM tiles
    __shared__ __align__(64) __bf16 xs[2][T_SEQ * 32];   // 2 x  8KB bf16 tiles

    const int tid  = threadIdx.x;        // 0..127
    const int wv   = tid >> 5;           // 0..3
    const int lane = tid & 31;
    const int hf   = lane >> 4;
    const int l16  = lane & 15;
    const int nBase = blockIdx.x * 64 + wv * 16;

    v8f acc[8];
#pragma unroll
    for (int i = 0; i < 8; ++i) acc[i] = zero8();

    const float* wrow = W + (size_t)(nBase + l16) * D_MODEL + hf * 16;
    const unsigned ldsRaw0 = (unsigned)(uintptr_t)&xraw[0][0];
    const unsigned ldsRaw1 = (unsigned)(uintptr_t)&xraw[1][0];

    if (wv == 0)                                   // wave0 drives the TDM pipe
        tdm_load_2d_f32(X, ldsRaw0, 32, T_SEQ, D_MODEL, T_SEQ, D_MODEL);

    for (int k0 = 0; k0 < D_MODEL; k0 += 32) {
        const int buf = (k0 >> 5) & 1;
        if (wv == 0) wait_tensorcnt0();            // raw[buf] landed
        __syncthreads();                           // publish raw[buf] to all waves
        if (wv == 0 && k0 + 32 < D_MODEL)          // kick next tile while we work
            tdm_load_2d_f32(X + k0 + 32, buf ? ldsRaw0 : ldsRaw1,
                            32, T_SEQ, D_MODEL, T_SEQ, D_MODEL);

        // ---- LDS fp32 -> LDS bf16 conversion pass (row tid) ----
        {
            const float4* src = (const float4*)&xraw[buf][tid * 32];
            float xb[32];
#pragma unroll
            for (int j = 0; j < 8; ++j) ((float4*)xb)[j] = src[j];
            __bf16* dst = &xs[buf][tid * 32];
#pragma unroll
            for (int i = 0; i < 32; ++i) dst[i] = (__bf16)xb[i];
        }

        // ---- B fragment: 16 W rows x 32 K, fp32 -> bf16 in VGPRs ----
        const float4* wp = (const float4*)(wrow + k0);
        __builtin_prefetch(wrow + k0 + 64, 0, 1);     // global_prefetch_b8
        float wb[16];
        ((float4*)wb)[0] = wp[0];
        ((float4*)wb)[1] = wp[1];
        ((float4*)wb)[2] = wp[2];
        ((float4*)wb)[3] = wp[3];
        v16bf bfrag;
#pragma unroll
        for (int i = 0; i < 16; ++i) bfrag[i] = (__bf16)wb[i];

        __syncthreads();                           // xs[buf] ready

        // ---- preload all 8 A fragments, then dense WMMA stream ----
        v16bf af[8];
#pragma unroll
        for (int ms = 0; ms < 8; ++ms)
            af[ms] = *(const v16bf*)&xs[buf][(ms * 16 + l16) * 32 + hf * 16];
        SCHED_FENCE();
#pragma unroll
        for (int ms = 0; ms < 8; ++ms)
            acc[ms] = __builtin_amdgcn_wmma_f32_16x16x32_bf16(
                false, af[ms], false, bfrag, (short)0, acc[ms], false, false);
    }

    const float b = bias[nBase + l16];
#pragma unroll
    for (int ms = 0; ms < 8; ++ms) {
#pragma unroll
        for (int r = 0; r < 8; ++r) {
            const int m = ms * 16 + hf * 8 + r;   // C layout: VGPR r, lane-half
            C[(size_t)m * D_MODEL + nBase + l16] = acc[ms][r] + b;
        }
    }
}

// ---------------------------------------------------------------------------
// Precompute RoPE cos/sin for the single scalar position (128 angles).
// ---------------------------------------------------------------------------
__global__ void rope_trig_kernel(const int* __restrict__ posP, float* __restrict__ trig)
{
    const int hd = threadIdx.x;               // 0..127
    const float pos = (float)(*posP);
    const int i = hd & 63;
    const float ang = pos * __powf(10000.0f, -(float)(2 * i) / 128.0f);
    trig[hd]       = __cosf(ang);
    trig[128 + hd] = __sinf(ang);
}

// ---------------------------------------------------------------------------
// Flash-decoding attention: grid = (32 heads, 4 S-chunks), block = 256 thr
// (8 waves, one 16-row query tile each). K/V staged per 32-key step into
// double-buffered LDS (RoPE applied to K on the fly), shared by all waves.
// Emits UNNORMALIZED partial O plus per-row (m, l); combined afterwards.
// ---------------------------------------------------------------------------
__global__ __launch_bounds__(256)
void flash_attn_kernel(const float* __restrict__ qRaw,
                       const float* __restrict__ kCache,
                       const float* __restrict__ vCache,
                       const float* __restrict__ kNew,
                       const float* __restrict__ vNew,
                       const float* __restrict__ trig,
                       const int*   __restrict__ cachePosP,
                       float* __restrict__ oPart,   // [4][T*D_MODEL]
                       float* __restrict__ mPart,   // [4][H*T]
                       float* __restrict__ lPart)   // [4][H*T]
{
    __shared__ __align__(64) __bf16 qs[T_SEQ * HEAD_DIM];    // 32 KB rotated Q
    __shared__ __align__(64) __bf16 ks[2][32 * HEAD_DIM];    // 2 x 8 KB rotated K
    __shared__ __align__(64) __bf16 vs[2][32 * HEAD_DIM];    // 2 x 8 KB
    __shared__ __align__(64) __bf16 ps[8 * 16 * 32];         // 8 KB P redistribute
    __shared__ float trigS[256];

    const int head  = blockIdx.x;
    const int chunk = blockIdx.y;
    const int tid   = threadIdx.x;
    const int wv    = tid >> 5;
    const int lane  = tid & 31;
    const int hf    = lane >> 4;
    const int l16   = lane & 15;

    trigS[tid] = trig[tid];
    const int cpos = *cachePosP;
    const int S    = cpos + T_SEQ;
    const int cl   = ((S + N_CHUNK * 32 - 1) / (N_CHUNK * 32)) * 32;
    const int sStart = chunk * cl;
    const int sEnd   = (sStart + cl < S) ? (sStart + cl) : S;
    __syncthreads();

    // ---- stage rotated Q (fp32 -> bf16): thread -> row tid>>1, 64-wide half
    {
        const int row = tid >> 1;
        const int hb  = (tid & 1) * 64;
        const float* qp = qRaw + (size_t)row * D_MODEL + head * HEAD_DIM;
        for (int i = 0; i < 64; ++i) {
            const int hd  = hb + i;
            const int hd2 = hd ^ 64;
            const float sgn = (hd < 64) ? -1.0f : 1.0f;
            const float v = qp[hd] * trigS[hd] + sgn * qp[hd2] * trigS[128 + hd];
            qs[row * HEAD_DIM + hd] = (__bf16)v;
        }
    }
    __syncthreads();

    v16bf qfrag[4];
#pragma unroll
    for (int c = 0; c < 4; ++c)
        qfrag[c] = *(const v16bf*)&qs[(wv * 16 + l16) * HEAD_DIM + c * 32 + hf * 16];

    v8f oacc[8];
#pragma unroll
    for (int i = 0; i < 8; ++i) oacc[i] = zero8();
    float mrow[8], lrow[8];
#pragma unroll
    for (int r = 0; r < 8; ++r) { mrow[r] = -1e30f; lrow[r] = 0.0f; }
    const float scale = 0.08838834764831845f;   // 1/sqrt(128)

    // stage 32 keys: thread -> key tid>>3, 16-wide hd eighth; clamp at sEnd
    auto stageKV = [&](int sBase, int buf) {
        const int key = tid >> 3;
        const int hb  = (tid & 7) * 16;
        int sIdx = sBase + key;
        sIdx = (sIdx < sEnd - 1) ? sIdx : (sEnd - 1);
        const float* kp = (sIdx < cpos)
            ? (kCache + (size_t)sIdx * D_MODEL + head * HEAD_DIM)
            : (kNew   + (size_t)(sIdx - cpos) * D_MODEL + head * HEAD_DIM);
#pragma unroll
        for (int i = 0; i < 16; ++i) {
            const int hd  = hb + i;
            const int hd2 = hd ^ 64;
            const float sgn = (hd < 64) ? -1.0f : 1.0f;
            const float v = kp[hd] * trigS[hd] + sgn * kp[hd2] * trigS[128 + hd];
            ks[buf][key * HEAD_DIM + hd] = (__bf16)v;
        }
        const float* vp = (sIdx < cpos)
            ? (vCache + (size_t)sIdx * D_MODEL + head * HEAD_DIM)
            : (vNew   + (size_t)(sIdx - cpos) * D_MODEL + head * HEAD_DIM);
        const float4* vp4 = (const float4*)(vp + hb);
        float vb[16];
        ((float4*)vb)[0] = vp4[0];
        ((float4*)vb)[1] = vp4[1];
        ((float4*)vb)[2] = vp4[2];
        ((float4*)vb)[3] = vp4[3];
        __bf16* vdst = &vs[buf][key * HEAD_DIM + hb];
#pragma unroll
        for (int i = 0; i < 16; ++i) vdst[i] = (__bf16)vb[i];
    };

    if (sStart < sEnd) stageKV(sStart, 0);

    int it = 0;
    for (int sBase = sStart; sBase < sEnd; sBase += 32, ++it) {
        const int buf = it & 1;
        __syncthreads();                        // tile[buf] staged, prev reads done
        if (sBase + 32 < sEnd) stageKV(sBase + 32, buf ^ 1);

        // ---- scores: Q[16x128] . K^T, two 16-key tiles, 4 hd chunks ----
        v16bf kf[8];
#pragma unroll
        for (int c = 0; c < 4; ++c) {
            kf[2 * c]     = *(const v16bf*)&ks[buf][l16 * HEAD_DIM + c * 32 + hf * 16];
            kf[2 * c + 1] = *(const v16bf*)&ks[buf][(16 + l16) * HEAD_DIM + c * 32 + hf * 16];
        }
        SCHED_FENCE();
        v8f sc0 = zero8(), sc1 = zero8();
#pragma unroll
        for (int c = 0; c < 4; ++c) {
            sc0 = __builtin_amdgcn_wmma_f32_16x16x32_bf16(
                false, qfrag[c], false, kf[2 * c],     (short)0, sc0, false, false);
            sc1 = __builtin_amdgcn_wmma_f32_16x16x32_bf16(
                false, qfrag[c], false, kf[2 * c + 1], (short)0, sc1, false, false);
        }

        // ---- online softmax over the 32 new keys (mask tail) ----
        const bool val0 = (sBase + l16)      < sEnd;
        const bool val1 = (sBase + 16 + l16) < sEnd;
        float p0[8], p1[8];
#pragma unroll
        for (int r = 0; r < 8; ++r) {
            const float s0v = val0 ? sc0[r] * scale : -1e30f;
            const float s1v = val1 ? sc1[r] * scale : -1e30f;
            const float tmax = half_max16(fmaxf(s0v, s1v));
            const float mNew = fmaxf(mrow[r], tmax);
            const float corr = __expf(mrow[r] - mNew);
            const float e0 = __expf(s0v - mNew);
            const float e1 = __expf(s1v - mNew);
            lrow[r] = lrow[r] * corr + half_sum16(e0 + e1);
            mrow[r] = mNew;
            p0[r] = e0; p1[r] = e1;
#pragma unroll
            for (int c = 0; c < 8; ++c) oacc[c][r] *= corr;
        }

        // ---- redistribute P (C layout -> A layout) through per-wave LDS ----
        __bf16* pw = &ps[wv * 512];
#pragma unroll
        for (int r = 0; r < 8; ++r) {
            pw[(hf * 8 + r) * 32 + l16]      = (__bf16)p0[r];
            pw[(hf * 8 + r) * 32 + 16 + l16] = (__bf16)p1[r];
        }
        v16bf pfrag = *(const v16bf*)&pw[l16 * 32 + hf * 16];  // same-wave DS order

        // ---- O += P[16x32] @ V[32x128], 8 hd chunks ----
        v16bf vf[8];
#pragma unroll
        for (int c = 0; c < 8; ++c)
#pragma unroll
            for (int s = 0; s < 16; ++s)
                vf[c][s] = vs[buf][(hf * 16 + s) * HEAD_DIM + c * 16 + l16];
        SCHED_FENCE();
#pragma unroll
        for (int c = 0; c < 8; ++c)
            oacc[c] = __builtin_amdgcn_wmma_f32_16x16x32_bf16(
                false, pfrag, false, vf[c], (short)0, oacc[c], false, false);
    }

    // ---- write unnormalized partial O and per-row stats ----
    float* oP = oPart + (size_t)chunk * (T_SEQ * D_MODEL);
#pragma unroll
    for (int r = 0; r < 8; ++r) {
        const int t = wv * 16 + hf * 8 + r;
#pragma unroll
        for (int c = 0; c < 8; ++c)
            oP[(size_t)t * D_MODEL + head * HEAD_DIM + c * 16 + l16] = oacc[c][r];
        if (l16 == 0) {
            mPart[chunk * (32 * T_SEQ) + head * T_SEQ + t] = mrow[r];
            lPart[chunk * (32 * T_SEQ) + head * T_SEQ + t] = lrow[r];
        }
    }
}

// ---------------------------------------------------------------------------
// Merge the 4 flash-decoding partials: out = sum_c w_c*O_c / sum_c w_c*l_c,
// w_c = exp(m_c - max_c m_c). One thread per output element.
// ---------------------------------------------------------------------------
__global__ __launch_bounds__(256)
void attn_combine_kernel(const float* __restrict__ oPart,
                         const float* __restrict__ mPart,
                         const float* __restrict__ lPart,
                         float* __restrict__ attnO)
{
    const int idx = blockIdx.x * 256 + threadIdx.x;   // < 128*4096
    const int t    = idx >> 12;
    const int col  = idx & 4095;
    const int head = col >> 7;
    const int mlIdx = head * T_SEQ + t;

    float m[4];
#pragma unroll
    for (int c = 0; c < N_CHUNK; ++c) m[c] = mPart[c * (32 * T_SEQ) + mlIdx];
    const float mG = fmaxf(fmaxf(m[0], m[1]), fmaxf(m[2], m[3]));
    float acc = 0.f, den = 0.f;
#pragma unroll
    for (int c = 0; c < N_CHUNK; ++c) {
        const float w = __expf(m[c] - mG);
        acc += w * oPart[(size_t)c * (T_SEQ * D_MODEL) + idx];
        den += w * lPart[c * (32 * T_SEQ) + mlIdx];
    }
    attnO[idx] = acc / den;
}

// ---------------------------------------------------------------------------
extern "C" void kernel_launch(void* const* d_in, const int* in_sizes, int n_in,
                              void* d_out, int out_size, void* d_ws, size_t ws_size,
                              hipStream_t stream)
{
    (void)in_sizes; (void)n_in; (void)out_size; (void)ws_size;

    const float* x    = (const float*)d_in[0];
    const float* wq_w = (const float*)d_in[1];
    const float* wq_b = (const float*)d_in[2];
    const float* wk_w = (const float*)d_in[3];
    const float* wk_b = (const float*)d_in[4];
    const float* wv_w = (const float*)d_in[5];
    const float* wv_b = (const float*)d_in[6];
    const float* wo_w = (const float*)d_in[7];
    const float* wo_b = (const float*)d_in[8];
    const float* kC   = (const float*)d_in[9];
    const float* vC   = (const float*)d_in[10];
    const int*   pos  = (const int*)d_in[11];
    const int*   cpos = (const int*)d_in[12];

    const size_t MB2 = (size_t)T_SEQ * D_MODEL * sizeof(float);   // 2 MB
    char*  ws    = (char*)d_ws;
    float* trig  = (float*)(ws);                          // 4 KB slot
    float* qRaw  = (float*)(ws + 4096);
    float* kNew  = (float*)(ws + 4096 + 1 * MB2);
    float* vNew  = (float*)(ws + 4096 + 2 * MB2);
    float* attnO = (float*)(ws + 4096 + 3 * MB2);
    float* oPart = (float*)(ws + 4096 + 4 * MB2);         // 4 x 2 MB
    float* mPart = (float*)(ws + 4096 + 8 * MB2);         // 64 KB
    float* lPart = (float*)(ws + 4096 + 8 * MB2 + (N_CHUNK * 32 * T_SEQ) * sizeof(float));

    rope_trig_kernel<<<dim3(1), dim3(128), 0, stream>>>(pos, trig);
    gemm_xwT_bias<<<dim3(64), dim3(128), 0, stream>>>(x, wq_w, wq_b, qRaw);
    gemm_xwT_bias<<<dim3(64), dim3(128), 0, stream>>>(x, wk_w, wk_b, kNew);
    gemm_xwT_bias<<<dim3(64), dim3(128), 0, stream>>>(x, wv_w, wv_b, vNew);
    flash_attn_kernel<<<dim3(32, N_CHUNK), dim3(256), 0, stream>>>(
        qRaw, kC, vC, kNew, vNew, trig, cpos, oPart, mPart, lPart);
    attn_combine_kernel<<<dim3((T_SEQ * D_MODEL) / 256), dim3(256), 0, stream>>>(
        oPart, mPart, lPart, attnO);
    gemm_xwT_bias<<<dim3(64), dim3(128), 0, stream>>>(attnO, wo_w, wo_b, (float*)d_out);
}
